// SparseLinear_2113123910054
// MI455X (gfx1250) — matrix operand order repaired and checked by reading, not compiled, not run on hardware
//
#include <hip/hip_runtime.h>

// ---------- types for WMMA fragments (gfx1250, wave32) ----------
typedef _Float16 v16h __attribute__((ext_vector_type(16)));
typedef _Float16 v8h  __attribute__((ext_vector_type(8)));
typedef float    v8f  __attribute__((ext_vector_type(8)));

#define D           512          // d_in == d_out
#define ROWS_TOTAL  65536
#define RPB         16           // rows per bin (one 16-row WMMA block)
#define BINS        (ROWS_TOTAL / RPB)   // 4096

// ------------------------------------------------------------------
// Kernel 0: convert W (f32 [d_out][d_in]) to f16 in identical layout.
// That layout IS the B-matrix (B[k][n] = W[n][k]) read pattern we need.
// Also zero the bin counters.
// ------------------------------------------------------------------
__global__ void k_prep(const float* __restrict__ w, _Float16* __restrict__ wh,
                       unsigned* __restrict__ counts) {
  int i = blockIdx.x * blockDim.x + threadIdx.x;
  if (i < D * D) wh[i] = (_Float16)w[i];
  if (i < BINS)  counts[i] = 0u;
}

// ------------------------------------------------------------------
// Kernel 1: histogram nonzeros into 16-row bins
// ------------------------------------------------------------------
__global__ void k_count(const int* __restrict__ rows,
                        unsigned* __restrict__ counts, int nnz) {
  int i = blockIdx.x * blockDim.x + threadIdx.x;
  if (i < nnz) atomicAdd(&counts[((unsigned)rows[i]) >> 4], 1u);
}

// ------------------------------------------------------------------
// Kernel 2: exclusive scan of 4096 counters. One block, 1024 threads,
// 4 bins/thread; Hillis-Steele over thread partials in LDS.
// ------------------------------------------------------------------
__global__ void k_scan(const unsigned* __restrict__ counts,
                       unsigned* __restrict__ offsets,
                       unsigned* __restrict__ cursor) {
  __shared__ unsigned s[1024];
  int t = threadIdx.x;
  unsigned c[4];
  unsigned sum = 0u;
  for (int k = 0; k < 4; ++k) { c[k] = counts[t * 4 + k]; sum += c[k]; }
  s[t] = sum;
  __syncthreads();
  for (int off = 1; off < 1024; off <<= 1) {
    unsigned v = (t >= off) ? s[t - off] : 0u;
    __syncthreads();
    s[t] += v;
    __syncthreads();
  }
  unsigned ex = (t > 0) ? s[t - 1] : 0u;
  for (int k = 0; k < 4; ++k) {
    offsets[t * 4 + k] = ex;
    cursor [t * 4 + k] = ex;
    ex += c[k];
  }
  if (t == 1023) offsets[BINS] = ex;
}

// ------------------------------------------------------------------
// Kernel 3: scatter nonzeros into per-bin compact storage.
// packed = { value bits, (row&15)<<16 | col }
// ------------------------------------------------------------------
__global__ void k_scatter(const int* __restrict__ rows,
                          const int* __restrict__ cols,
                          const float* __restrict__ vals,
                          unsigned* __restrict__ cursor,
                          uint2* __restrict__ packed, int nnz) {
  int i = blockIdx.x * blockDim.x + threadIdx.x;
  if (i < nnz) {
    unsigned r   = (unsigned)rows[i];
    unsigned pos = atomicAdd(&cursor[r >> 4], 1u);
    uint2 e;
    e.x = __float_as_uint(vals[i]);
    e.y = ((r & 15u) << 16) | (unsigned)cols[i];
    packed[pos] = e;
  }
}

// ------------------------------------------------------------------
// Kernel 4: per-bin dense 16x512x512 GEMM with v_wmma_f32_16x16x32_f16.
//  - densify the bin's nonzeros into a 16x512 f32 LDS tile (ds_add_f32)
//  - convert to f16 in LDS
//  - 8 waves x 4 N-tiles, K in steps of 32, f32 accumulators
//  - store with bias (every row written -> empty rows get bias, as ref)
// ------------------------------------------------------------------
__global__ void __launch_bounds__(256)
k_gemm(const uint2* __restrict__ packed,
       const unsigned* __restrict__ offsets,
       const _Float16* __restrict__ wh,
       const float* __restrict__ bias,
       float* __restrict__ out) {
  __shared__ __align__(32) float    Af[RPB * D];   // 32 KB f32 accumulation tile
  __shared__ __align__(32) _Float16 Ah[RPB * D];   // 16 KB f16 A tile

  const int bin = blockIdx.x;
  const int tid = threadIdx.x;

  // zero the accumulation tile
  for (int i = tid; i < RPB * D; i += 256) Af[i] = 0.0f;
  __syncthreads();

  // scatter-accumulate this bin's nonzeros (duplicates handled by ds_add_f32)
  const unsigned beg = offsets[bin], end = offsets[bin + 1];
  for (unsigned i = beg + tid; i < end; i += 256) {
    uint2 e = packed[i];
    unsigned rl = e.y >> 16;
    unsigned c  = e.y & 0xFFFFu;
    atomicAdd(&Af[rl * D + c], __uint_as_float(e.x));
  }
  __syncthreads();

  // convert to f16 for the matrix core
  for (int i = tid; i < RPB * D; i += 256) Ah[i] = (_Float16)Af[i];
  __syncthreads();

  const int lane = tid & 31;
  const int wave = tid >> 5;
  const int M    = lane & 15;       // row within 16-row block (A) / col (B,D)
  const int hi   = lane >> 4;       // lane-group select per ISA layouts
  const int n0   = wave * 64;       // each wave owns 4 tiles of 16 output cols

  v8f acc[4] = {v8f{}, v8f{}, v8f{}, v8f{}};

  for (int k0 = 0; k0 < D; k0 += 32) {
    // A fragment (16x32, contiguous k order, same as B): lane row M,
    // halves j=0..7  -> K = k0 + hi*16 + j
    // halves j=8..15 -> K = k0 + hi*16 + j
    v8h lo = *(const v8h*)&Ah[M * D + k0 + hi * 16];
    v8h hh = *(const v8h*)&Ah[M * D + k0 + hi * 16 + 8];
    v16h a;
    #pragma unroll
    for (int j = 0; j < 8; ++j) { a[j] = lo[j]; a[j + 8] = hh[j]; }

    // B fragment (32x16): lane col N = tile + M, K = k0 + hi*16 + j
    const int kb = k0 + hi * 16;
    #pragma unroll
    for (int tt = 0; tt < 4; ++tt) {
      const int N = n0 + tt * 16 + M;
      v16h b = *(const v16h*)(wh + N * D + kb);   // 32B contiguous, L2-resident
      acc[tt] = __builtin_amdgcn_wmma_f32_16x16x32_f16(
          /*neg_a=*/false, a, /*neg_b=*/false, b,
          /*c_mod=*/(short)0, acc[tt], /*reuse_a=*/false, /*reuse_b=*/false);
    }
  }

  // D layout: N = lane&15, VGPR i -> row M = i + 8*hi.  Add bias and store.
  const int rowbase = bin * RPB + hi * 8;
  #pragma unroll
  for (int tt = 0; tt < 4; ++tt) {
    const int n = n0 + tt * 16 + M;
    const float bb = bias[n];
    #pragma unroll
    for (int i = 0; i < 8; ++i) {
      out[(size_t)(rowbase + i) * D + n] = acc[tt][i] + bb;
    }
  }
}

// ------------------------------------------------------------------
extern "C" void kernel_launch(void* const* d_in, const int* in_sizes, int n_in,
                              void* d_out, int out_size, void* d_ws, size_t ws_size,
                              hipStream_t stream) {
  const int*   rows = (const int*)d_in[0];
  const int*   cols = (const int*)d_in[1];
  const float* vals = (const float*)d_in[2];
  const float* w    = (const float*)d_in[3];
  const float* bias = (const float*)d_in[4];
  float*       out  = (float*)d_out;
  const int nnz = in_sizes[0];

  // workspace layout (~4.6 MB, base assumed >=256B aligned)
  char* ws = (char*)d_ws;
  _Float16* wh      = (_Float16*)ws;                              // 512 KB
  unsigned* counts  = (unsigned*)(ws + 524288);                   // 16 KB
  unsigned* offsets = (unsigned*)(ws + 524288 + 16384);           // 16388 B (padded)
  unsigned* cursor  = (unsigned*)(ws + 524288 + 16384 + 16640);   // 16 KB
  uint2*    packed  = (uint2*)   (ws + 524288 + 16384 + 16640 + 16384); // nnz*8 B

  k_prep   <<<(D * D + 255) / 256, 256, 0, stream>>>(w, wh, counts);
  k_count  <<<(nnz + 255) / 256,   256, 0, stream>>>(rows, counts, nnz);
  k_scan   <<<1, 1024, 0, stream>>>(counts, offsets, cursor);
  k_scatter<<<(nnz + 255) / 256,   256, 0, stream>>>(rows, cols, vals, cursor, packed, nnz);
  k_gemm   <<<BINS, 256, 0, stream>>>(packed, offsets, wh, bias, out);
}
